// TrianDecoder_28552942583888
// MI455X (gfx1250) — matrix operand order repaired and checked by reading, not compiled, not run on hardware
//
#include <hip/hip_runtime.h>
#include <hip/hip_bf16.h>

typedef __bf16 bf16_t;
typedef __attribute__((ext_vector_type(16))) __bf16 v16bf;
typedef __attribute__((ext_vector_type(8)))  __bf16 v8bf;
typedef __attribute__((ext_vector_type(8)))  float  v8f;

#define NLAYERS 4
#define BB 32
#define TT 64
#define HH 1024
#define VV 32000
#define GG 4096      /* 4*H */

// ---------- helpers ----------
__device__ __forceinline__ bf16_t f2bf(float x) {
    unsigned u = __builtin_bit_cast(unsigned, x);
    u += 0x7fffu + ((u >> 16) & 1u);               // round-to-nearest-even
    unsigned short h = (unsigned short)(u >> 16);
    return __builtin_bit_cast(bf16_t, h);
}

__device__ __forceinline__ v8f wmma_bf16(v16bf a, v16bf b, v8f c) {
    // D = A(16x32 bf16) * B(32x16 bf16) + C(16x16 f32)
    return __builtin_amdgcn_wmma_f32_16x16x32_bf16(
        false, a, false, b, (short)0, c, false, false);
}

// A fragment: lane L holds row m=L&15; lanes0-15: K{0..15}, lanes16-31: K{16..31}
// p points at row base + half*16 elements; loads p[0..7] and p[8..15].
__device__ __forceinline__ v16bf load_afrag(const bf16_t* p) {
    union { v16bf v; v8bf h[2]; } u;
    u.h[0] = *(const v8bf*)(p);
    u.h[1] = *(const v8bf*)(p + 8);
    return u.v;
}

// ---------- conversion / staging kernels ----------

// out[n*ldo + k] = bf16(in[k*N + n]);  in: K x N fp32 row-major
__global__ __launch_bounds__(256) void k_transpose_convert(
    const float* __restrict__ in, int K, int N, bf16_t* __restrict__ out, int ldo) {
    __shared__ float tile[32][33];
    int n0 = blockIdx.x * 32, k0 = blockIdx.y * 32;
    int tx = threadIdx.x, ty = threadIdx.y;
    for (int i = ty; i < 32; i += 8)
        tile[i][tx] = in[(size_t)(k0 + i) * N + n0 + tx];
    __syncthreads();
    for (int i = ty; i < 32; i += 8)
        out[(size_t)(n0 + i) * ldo + k0 + tx] = f2bf(tile[tx][i]);
}

// xseq[(t*32+b)*1024 + e] = bf16(emb[tok[b*64+t]][e])
__global__ __launch_bounds__(256) void k_gather_embed(
    const int* __restrict__ tok, const float* __restrict__ emb, bf16_t* __restrict__ xseq) {
    int bx = blockIdx.x;                 // 0..2047 = t*32+b
    int t = bx >> 5, b = bx & 31;
    int vtok = tok[b * TT + t];
    const float* src = emb + (size_t)vtok * HH;
    bf16_t* dst = xseq + (size_t)bx * HH;
    for (int e = threadIdx.x; e < HH; e += 256) dst[e] = f2bf(src[e]);
}

// hbuf bf16 <- states[0::2]; cbuf fp32 <- states[1::2]
__global__ __launch_bounds__(256) void k_init_state(
    const float* __restrict__ states, bf16_t* __restrict__ hbuf, float* __restrict__ cbuf) {
    int idx = blockIdx.x * 256 + threadIdx.x;     // < 4*32*1024
    int l = idx >> 15, r = idx & 32767;
    hbuf[idx] = f2bf(states[(size_t)(2 * l) * 32768 + r]);
    cbuf[idx] = states[(size_t)(2 * l + 1) * 32768 + r];
}

// ---------- LSTM step GEMM: Z = [Ax;Ah](32x2048) * BT^T + bias ----------
// BT: (4096 x 2048) bf16, N-major (row n = K contiguous). Grid 64 x 128thr (4 waves).
__global__ __launch_bounds__(128) void k_lstm_gemm(
    const bf16_t* __restrict__ Ax, const bf16_t* __restrict__ Ah,
    const bf16_t* __restrict__ BT, const float* __restrict__ bias,
    float* __restrict__ Z) {
    int lane = threadIdx.x & 31;
    int w    = threadIdx.x >> 5;
    int lane15 = lane & 15;
    int half   = lane >> 4;
    int nbase  = blockIdx.x * 64 + w * 16;

    v8f acc0 = {};   // rows 0..15
    v8f acc1 = {};   // rows 16..31

    const bf16_t* bp = BT + (size_t)(nbase + lane15) * 2048 + half * 16;

    #pragma unroll
    for (int part = 0; part < 2; ++part) {
        const bf16_t* A = part ? Ah : Ax;
        const bf16_t* a0 = A + (size_t)lane15 * HH + half * 16;
        const bf16_t* a1 = A + (size_t)(16 + lane15) * HH + half * 16;
        const bf16_t* bpp = bp + part * HH;
        #pragma unroll 4
        for (int k = 0; k < HH; k += 32) {
            v16bf af0 = load_afrag(a0 + k);
            v16bf af1 = load_afrag(a1 + k);
            v16bf bf_ = *(const v16bf*)(bpp + k);
            acc0 = wmma_bf16(af0, bf_, acc0);
            acc1 = wmma_bf16(af1, bf_, acc1);
        }
    }

    int n = nbase + lane15;
    float bv = bias[n];
    int mrow = half * 8;
    #pragma unroll
    for (int r = 0; r < 8; ++r) {
        Z[(size_t)(mrow + r) * GG + n]        = acc0[r] + bv;
        Z[(size_t)(16 + mrow + r) * GG + n]   = acc1[r] + bv;
    }
}

// ---------- LSTM gates ----------
__global__ __launch_bounds__(256) void k_lstm_gates(
    const float* __restrict__ Z, float* __restrict__ C,
    bf16_t* __restrict__ Hout, bf16_t* __restrict__ Top) {
    int idx = blockIdx.x * 256 + threadIdx.x;    // < 32768
    int m = idx >> 10, j = idx & 1023;
    const float* zr = Z + (size_t)m * GG;
    float zi = zr[j], zf = zr[HH + j], zg = zr[2 * HH + j], zo = zr[3 * HH + j];
    float si = 1.0f / (1.0f + __expf(-zi));
    float sf = 1.0f / (1.0f + __expf(-zf));
    float so = 1.0f / (1.0f + __expf(-zo));
    float c  = sf * C[idx] + si * tanhf(zg);
    C[idx] = c;
    float h = so * tanhf(c);
    bf16_t hb = f2bf(h);
    Hout[idx] = hb;
    if (Top) Top[idx] = hb;
}

// ---------- FC GEMM: logits = tops(2048x1024) * WfcT^T + bfc, scattered to (b,t,v) ----------
// grid (32, 250) x 128thr (4 waves); each wave: 64 rows x 32 cols = 8 WMMA tiles.
// Per K-step: 4 A-frags + 2 B-frags feed 8 WMMAs (0.75 loads/WMMA), halving L2 traffic
// for both A (250 vs 500 col-WGs) and Wfc (32 vs 64 row passes) vs a 32x16 tiling.
__global__ __launch_bounds__(128) void k_fc_gemm(
    const bf16_t* __restrict__ A, const bf16_t* __restrict__ BT,
    const float* __restrict__ bfc, float* __restrict__ out) {
    int lane = threadIdx.x & 31;
    int w    = threadIdx.x >> 5;
    int lane15 = lane & 15;
    int half   = lane >> 4;
    int m0     = blockIdx.x * 64;                   // 64 rows per WG
    int nbase  = (blockIdx.y * 4 + w) * 32;         // 32 cols per wave

    v8f acc[4][2];
    #pragma unroll
    for (int mt = 0; mt < 4; ++mt)
        #pragma unroll
        for (int nt = 0; nt < 2; ++nt) acc[mt][nt] = (v8f){};

    const bf16_t* ap[4];
    #pragma unroll
    for (int mt = 0; mt < 4; ++mt)
        ap[mt] = A + (size_t)(m0 + mt * 16 + lane15) * HH + half * 16;
    const bf16_t* bp[2];
    #pragma unroll
    for (int nt = 0; nt < 2; ++nt)
        bp[nt] = BT + (size_t)(nbase + nt * 16 + lane15) * HH + half * 16;

    #pragma unroll 2
    for (int k = 0; k < HH; k += 32) {
        v16bf af[4], bf_[2];
        #pragma unroll
        for (int mt = 0; mt < 4; ++mt) af[mt] = load_afrag(ap[mt] + k);
        #pragma unroll
        for (int nt = 0; nt < 2; ++nt) bf_[nt] = *(const v16bf*)(bp[nt] + k);
        #pragma unroll
        for (int mt = 0; mt < 4; ++mt)
            #pragma unroll
            for (int nt = 0; nt < 2; ++nt)
                acc[mt][nt] = wmma_bf16(af[mt], bf_[nt], acc[mt][nt]);
    }

    int mrow = half * 8;
    #pragma unroll
    for (int nt = 0; nt < 2; ++nt) {
        int v = nbase + nt * 16 + lane15;
        float bv = bfc[v];
        #pragma unroll
        for (int mt = 0; mt < 4; ++mt) {
            #pragma unroll
            for (int r = 0; r < 8; ++r) {
                int m = m0 + mt * 16 + mrow + r;    // row of tops = t*32+b
                int t_ = m >> 5, b_ = m & 31;
                out[(size_t)(b_ * TT + t_) * VV + v] = acc[mt][nt][r] + bv;
            }
        }
    }
}

// ---------- row softmax over 32000, in place ----------
__global__ __launch_bounds__(256) void k_softmax(float* __restrict__ out) {
    __shared__ float red[256];
    int tid = threadIdx.x;
    float* p = out + (size_t)blockIdx.x * VV;
    float mx = -3.4e38f;
    for (int v = tid; v < VV; v += 256) mx = fmaxf(mx, p[v]);
    red[tid] = mx; __syncthreads();
    for (int s = 128; s > 0; s >>= 1) { if (tid < s) red[tid] = fmaxf(red[tid], red[tid + s]); __syncthreads(); }
    mx = red[0]; __syncthreads();
    float sum = 0.0f;
    for (int v = tid; v < VV; v += 256) { float e = __expf(p[v] - mx); p[v] = e; sum += e; }
    red[tid] = sum; __syncthreads();
    for (int s = 128; s > 0; s >>= 1) { if (tid < s) red[tid] += red[tid + s]; __syncthreads(); }
    float inv = 1.0f / red[0];
    __syncthreads();
    for (int v = tid; v < VV; v += 256) p[v] *= inv;
}

// ---------- host orchestration ----------
extern "C" void kernel_launch(void* const* d_in, const int* in_sizes, int n_in,
                              void* d_out, int out_size, void* d_ws, size_t ws_size,
                              hipStream_t stream) {
    const int*   tok    = (const int*)d_in[0];    // (32,64)
    const float* states = (const float*)d_in[1];  // (8,32,1024)
    const float* emb    = (const float*)d_in[2];  // (32000,1024)
    const float* Wx     = (const float*)d_in[3];  // (4,1024,4096)
    const float* Wh     = (const float*)d_in[4];  // (4,1024,4096)
    const float* bias   = (const float*)d_in[5];  // (4,4096)
    const float* Wfc    = (const float*)d_in[6];  // (1024,32000)
    const float* bfc    = (const float*)d_in[7];  // (32000,)
    float* out = (float*)d_out;                   // (32,64,32000)

    char* ws = (char*)d_ws;
    bf16_t* WxhT = (bf16_t*)ws;  ws += (size_t)NLAYERS * GG * 2048 * sizeof(bf16_t); // 67.1 MB
    bf16_t* WfcT = (bf16_t*)ws;  ws += (size_t)VV * HH * sizeof(bf16_t);             // 65.5 MB
    bf16_t* xseq = (bf16_t*)ws;  ws += (size_t)TT * BB * HH * sizeof(bf16_t);        // 4.2 MB
    bf16_t* tops = (bf16_t*)ws;  ws += (size_t)TT * BB * HH * sizeof(bf16_t);        // 4.2 MB
    bf16_t* hbuf = (bf16_t*)ws;  ws += (size_t)NLAYERS * BB * HH * sizeof(bf16_t);
    float*  cbuf = (float*)ws;   ws += (size_t)NLAYERS * BB * HH * sizeof(float);
    float*  zbuf = (float*)ws;   ws += (size_t)BB * GG * sizeof(float);

    // 1) weights -> bf16, transposed (N-major, K-contiguous)
    for (int l = 0; l < NLAYERS; ++l) {
        k_transpose_convert<<<dim3(GG / 32, HH / 32), dim3(32, 8), 0, stream>>>(
            Wx + (size_t)l * HH * GG, HH, GG, WxhT + (size_t)l * GG * 2048, 2048);
        k_transpose_convert<<<dim3(GG / 32, HH / 32), dim3(32, 8), 0, stream>>>(
            Wh + (size_t)l * HH * GG, HH, GG, WxhT + (size_t)l * GG * 2048 + HH, 2048);
    }
    k_transpose_convert<<<dim3(VV / 32, HH / 32), dim3(32, 8), 0, stream>>>(
        Wfc, HH, VV, WfcT, HH);

    // 2) embedding gather + convert, 3) init states
    k_gather_embed<<<TT * BB, 256, 0, stream>>>(tok, emb, xseq);
    k_init_state<<<(NLAYERS * BB * HH) / 256, 256, 0, stream>>>(states, hbuf, cbuf);

    // 4) recurrent chain
    for (int t = 0; t < TT; ++t) {
        for (int l = 0; l < NLAYERS; ++l) {
            const bf16_t* Ax = (l == 0) ? (xseq + (size_t)t * BB * HH)
                                        : (hbuf + (size_t)(l - 1) * BB * HH);
            const bf16_t* Ah = hbuf + (size_t)l * BB * HH;
            k_lstm_gemm<<<64, 128, 0, stream>>>(
                Ax, Ah, WxhT + (size_t)l * GG * 2048, bias + (size_t)l * GG, zbuf);
            bf16_t* top = (l == NLAYERS - 1) ? (tops + (size_t)t * BB * HH) : (bf16_t*)nullptr;
            k_lstm_gates<<<(BB * HH) / 256, 256, 0, stream>>>(
                zbuf, cbuf + (size_t)l * BB * HH, hbuf + (size_t)l * BB * HH, top);
        }
    }

    // 5) output head + 6) softmax
    k_fc_gemm<<<dim3(BB * TT / 64, VV / 128), 128, 0, stream>>>(tops, WfcT, bfc, out);
    k_softmax<<<TT * BB, 256, 0, stream>>>(out);
}
